// Dense_Warping_15367392985285
// MI455X (gfx1250) — matrix operand order repaired and checked by reading, not compiled, not run on hardware
//
#include <hip/hip_runtime.h>
#include <hip/hip_bf16.h>

#define HNUM 256
#define WNUM 256
#define HW   (HNUM * WNUM)
#define BNUM 2

typedef __attribute__((ext_vector_type(16))) __bf16       v16bf;
typedef __attribute__((ext_vector_type(8)))  float        v8f;
typedef __attribute__((ext_vector_type(4)))  unsigned int u32x4;

union FragU { u32x4 q[2]; v16bf v; };

__device__ __forceinline__ unsigned short f2bf(float f) {
  unsigned int u = __float_as_uint(f);
  u += 0x7fffu + ((u >> 16) & 1u);            // round-to-nearest-even
  return (unsigned short)(u >> 16);
}
__device__ __forceinline__ unsigned int f2bf2(float lo, float hi) {
  return (unsigned int)f2bf(lo) | ((unsigned int)f2bf(hi) << 16);
}

// A-fragment for v_wmma_*_16x16x32 (16-bit A, 16x32):
// lane<16 : row=lane,    K = 0..7  and 16..23
// lane>=16: row=lane-16, K = 8..15 and 24..31
__device__ __forceinline__ v16bf ldsAFrag(const unsigned short* rowBase, int lane) {
  FragU f;
  const int ko = (lane < 16) ? 0 : 8;
  f.q[0] = *(const u32x4*)(rowBase + ko);
  f.q[1] = *(const u32x4*)(rowBase + 16 + ko);
  return f.v;
}

// B-fragment (16-bit B, 32x16): lane<16 -> col=lane, K=0..7 and 16..23;
// lane>=16 -> col=lane-16, K=8..15 and 24..31. Weights packed as [ck][ocAll][32] bf16.
__device__ __forceinline__ v16bf glbBFrag(const unsigned short* __restrict__ wp,
                                          int ck, int ocAll, int ocBase, int lane) {
  const unsigned short* p = wp + (((size_t)ck * ocAll + ocBase + (lane & 15)) << 5)
                               + ((lane < 16) ? 0 : 8);
  FragU f;
  f.q[0] = *(const u32x4*)p;
  f.q[1] = *(const u32x4*)(p + 16);
  return f.v;
}

// ---------------------------------------------------------------- flow warp
__global__ __launch_bounds__(256) void warp_kernel(const float* __restrict__ nbr,
                                                   const float* __restrict__ flow,
                                                   float* __restrict__ nw) {
  int t = blockIdx.x * 256 + threadIdx.x;
  if (t >= BNUM * HW) return;
  int b = t / HW;
  int r = t - b * HW;
  int y = r >> 8;
  int x = r & 255;
  float fx = flow[(size_t)(b * 2 + 0) * HW + r];
  float fy = flow[(size_t)(b * 2 + 1) * HW + r];
  float px = (float)x + fx, py = (float)y + fy;
  float y0f = floorf(py), x0f = floorf(px);
  float wy = py - y0f, wx = px - x0f;
  int y0 = (int)y0f, x0 = (int)x0f, y1 = y0 + 1, x1 = x0 + 1;
  float vy0 = (y0 >= 0 && y0 < HNUM) ? 1.f : 0.f;
  float vy1 = (y1 >= 0 && y1 < HNUM) ? 1.f : 0.f;
  float vx0 = (x0 >= 0 && x0 < WNUM) ? 1.f : 0.f;
  float vx1 = (x1 >= 0 && x1 < WNUM) ? 1.f : 0.f;
  int cy0 = min(max(y0, 0), HNUM - 1), cy1 = min(max(y1, 0), HNUM - 1);
  int cx0 = min(max(x0, 0), WNUM - 1), cx1 = min(max(x1, 0), WNUM - 1);
  float w00 = (1.f - wy) * (1.f - wx) * vy0 * vx0;
  float w01 = (1.f - wy) * wx * vy0 * vx1;
  float w10 = wy * (1.f - wx) * vy1 * vx0;
  float w11 = wy * wx * vy1 * vx1;
  int i00 = cy0 * WNUM + cx0, i01 = cy0 * WNUM + cx1;
  int i10 = cy1 * WNUM + cx0, i11 = cy1 * WNUM + cx1;
  const float* src = nbr + (size_t)b * 64 * HW;
  float* dst = nw + (size_t)b * 64 * HW + r;
#pragma unroll 4
  for (int c = 0; c < 64; ++c) {
    const float* im = src + (size_t)c * HW;
    dst[(size_t)c * HW] = w00 * im[i00] + w01 * im[i01] + w10 * im[i10] + w11 * im[i11];
  }
}

// ------------------------------------------------------------ weight packs
// conv1: K ordering k = tap*128 + ic ; chunk ck -> tap=ck/4, icb=(ck&3)*32
__global__ __launch_bounds__(256) void pack_w1(const float* __restrict__ w,
                                               unsigned short* __restrict__ o, int n) {
  int e = blockIdx.x * 256 + threadIdx.x;
  if (e >= n) return;
  int ck = e >> 11;
  int oc = (e >> 5) & 63;
  int kk = e & 31;
  int tap = ck >> 2;
  int ic = ((ck & 3) << 5) + kk;
  o[e] = f2bf(w[((size_t)oc * 128 + ic) * 9 + tap]);
}
// om: ic padded 66->96 (3 chunks per tap), oc padded 216->256
__global__ __launch_bounds__(256) void pack_om(const float* __restrict__ w,
                                               unsigned short* __restrict__ o, int n) {
  int e = blockIdx.x * 256 + threadIdx.x;
  if (e >= n) return;
  int ck = e >> 13;
  int oc = (e >> 5) & 255;
  int kk = e & 31;
  int tap = ck / 3;
  int ic = (ck % 3) * 32 + kk;
  float v = 0.f;
  if (oc < 216 && ic < 66) v = w[((size_t)oc * 66 + ic) * 9 + tap];
  o[e] = f2bf(v);
}
// dcn: natural flatten k' = ic*9 + tap, 18 chunks
__global__ __launch_bounds__(256) void pack_dcn(const float* __restrict__ w,
                                                unsigned short* __restrict__ o, int n) {
  int e = blockIdx.x * 256 + threadIdx.x;
  if (e >= n) return;
  int ck = e >> 11;
  int oc = (e >> 5) & 63;
  int kk = e & 31;
  o[e] = f2bf(w[(size_t)oc * 576 + ck * 32 + kk]);
}

// ------------------------------------------------- conv1 (implicit GEMM + relu)
// Block = 128 thr = 4 waves. Tile: 32 pixels x 64 oc. Double-buffered A (2x32x32).
__global__ __launch_bounds__(128) void conv1_kernel(const float* __restrict__ nw,
                                                    const float* __restrict__ ref,
                                                    const unsigned short* __restrict__ w1p,
                                                    const float* __restrict__ bias,
                                                    float* __restrict__ fea) {
  __shared__ alignas(64) unsigned short At[2][32 * 32];
  const int xb = blockIdx.x * 32;
  const int y = blockIdx.y;
  const int b = blockIdx.z;
  const int lane = threadIdx.x & 31;
  const int wv = threadIdx.x >> 5;
  const int ocb = wv * 16;
  const int px = threadIdx.x & 31;
  const int icl = (threadIdx.x >> 5) << 3;   // 8 channels per thread

  auto fill = [&](int ck, int buf) {
    const int tap = ck >> 2;
    const int icb = (ck & 3) << 5;
    const int sy = y + tap / 3 - 1;
    const int sx = xb + px + tap % 3 - 1;
    const bool ok = (sy >= 0) & (sy < HNUM) & (sx >= 0) & (sx < WNUM);
#pragma unroll
    for (int j = 0; j < 8; j += 2) {
      const int ic = icb + icl + j;
      float v0 = 0.f, v1 = 0.f;
      if (ok) {
        const float* s0 = (ic < 64) ? (nw + (size_t)(b * 64 + ic) * HW)
                                    : (ref + (size_t)(b * 64 + ic - 64) * HW);
        const float* s1 = (ic + 1 < 64) ? (nw + (size_t)(b * 64 + ic + 1) * HW)
                                        : (ref + (size_t)(b * 64 + ic + 1 - 64) * HW);
        v0 = s0[sy * WNUM + sx];
        v1 = s1[sy * WNUM + sx];
      }
      *(unsigned int*)&At[buf][px * 32 + icl + j] = f2bf2(v0, v1);
    }
  };

  v8f acc0 = {}, acc1 = {};
  fill(0, 0);
  __syncthreads();
  for (int ck = 0; ck < 36; ++ck) {
    const int buf = ck & 1;
    v16bf a0 = ldsAFrag(At[buf] + (lane & 15) * 32, lane);
    v16bf a1 = ldsAFrag(At[buf] + ((lane & 15) + 16) * 32, lane);
    v16bf bv = glbBFrag(w1p, ck, 64, ocb, lane);
    acc0 = __builtin_amdgcn_wmma_f32_16x16x32_bf16(false, a0, false, bv,
                                                   (short)0, acc0, false, false);
    acc1 = __builtin_amdgcn_wmma_f32_16x16x32_bf16(false, a1, false, bv,
                                                   (short)0, acc1, false, false);
    if (ck + 1 < 36) fill(ck + 1, buf ^ 1);
    __syncthreads();
  }
  const int oc = ocb + (lane & 15);
  const float bb = bias[oc];
  const int m0 = (lane < 16) ? 0 : 8;
  float* dst = fea + ((size_t)(b * 64 + oc) * HNUM + y) * WNUM + xb + m0;
#pragma unroll
  for (int r = 0; r < 8; ++r) {
    float v0 = acc0[r] + bb;
    float v1 = acc1[r] + bb;
    dst[r]      = v0 > 0.f ? v0 : 0.f;
    dst[r + 16] = v1 > 0.f ? v1 : 0.f;
  }
}

// --------------------------------------------- offset/mask conv (implicit GEMM)
__global__ __launch_bounds__(128) void omconv_kernel(const float* __restrict__ fea,
                                                     const float* __restrict__ flow,
                                                     const unsigned short* __restrict__ wp,
                                                     const float* __restrict__ bias,
                                                     float* __restrict__ om) {
  __shared__ alignas(64) unsigned short At[2][32 * 32];
  const int ngrp = blockIdx.x & 3;
  const int xb = (blockIdx.x >> 2) * 32;
  const int y = blockIdx.y;
  const int b = blockIdx.z;
  const int lane = threadIdx.x & 31;
  const int wv = threadIdx.x >> 5;
  const int ocb = (ngrp * 4 + wv) * 16;   // 16 tiles x 16 = 256 padded oc
  const int px = threadIdx.x & 31;
  const int icl = (threadIdx.x >> 5) << 3;

  auto fill = [&](int ck, int buf) {
    const int tap = ck / 3;
    const int icb = (ck % 3) * 32;
    const int sy = y + tap / 3 - 1;
    const int sx = xb + px + tap % 3 - 1;
    const bool ok = (sy >= 0) & (sy < HNUM) & (sx >= 0) & (sx < WNUM);
#pragma unroll
    for (int j = 0; j < 8; j += 2) {
      const int ic = icb + icl + j;
      float v0 = 0.f, v1 = 0.f;
      if (ok) {
        if (ic < 64)          v0 = fea[(size_t)(b * 64 + ic) * HW + sy * WNUM + sx];
        else if (ic < 66)     v0 = flow[(size_t)(b * 2 + ic - 64) * HW + sy * WNUM + sx];
        if (ic + 1 < 64)      v1 = fea[(size_t)(b * 64 + ic + 1) * HW + sy * WNUM + sx];
        else if (ic + 1 < 66) v1 = flow[(size_t)(b * 2 + ic + 1 - 64) * HW + sy * WNUM + sx];
      }
      *(unsigned int*)&At[buf][px * 32 + icl + j] = f2bf2(v0, v1);
    }
  };

  v8f acc0 = {}, acc1 = {};
  fill(0, 0);
  __syncthreads();
  for (int ck = 0; ck < 27; ++ck) {
    const int buf = ck & 1;
    v16bf a0 = ldsAFrag(At[buf] + (lane & 15) * 32, lane);
    v16bf a1 = ldsAFrag(At[buf] + ((lane & 15) + 16) * 32, lane);
    v16bf bv = glbBFrag(wp, ck, 256, ocb, lane);
    acc0 = __builtin_amdgcn_wmma_f32_16x16x32_bf16(false, a0, false, bv,
                                                   (short)0, acc0, false, false);
    acc1 = __builtin_amdgcn_wmma_f32_16x16x32_bf16(false, a1, false, bv,
                                                   (short)0, acc1, false, false);
    if (ck + 1 < 27) fill(ck + 1, buf ^ 1);
    __syncthreads();
  }
  const int oc = ocb + (lane & 15);
  if (oc < 216) {
    const float bb = bias[oc];
    const int m0 = (lane < 16) ? 0 : 8;
    float* dst = om + ((size_t)b * 216 + oc) * HW + y * WNUM + xb + m0;
#pragma unroll
    for (int r = 0; r < 8; ++r) {
      dst[r]      = acc0[r] + bb;
      dst[r + 16] = acc1[r] + bb;
    }
  }
}

// -------------------------- deformable sampling + DCN reduction (WMMA einsum)
// Block = 128 thr = 4 waves. Tile: 32 pixels x 64 oc. A tile 32x576 bf16 = 36KB.
__global__ __launch_bounds__(128) void dcn_kernel(const float* __restrict__ nbr,
                                                  const float* __restrict__ flow,
                                                  const float* __restrict__ om,
                                                  const unsigned short* __restrict__ dcnp,
                                                  const float* __restrict__ bias,
                                                  float* __restrict__ out) {
  __shared__ alignas(64) unsigned short At[32 * 576];
  const int xb = blockIdx.x * 32;
  const int y = blockIdx.y;
  const int b = blockIdx.z;
  {
    const int p = threadIdx.x >> 2;     // pixel 0..31
    const int gq = threadIdx.x & 3;     // handles groups gq and gq+4
    const int x = xb + p;
    const int r = y * WNUM + x;
    const float* omb = om + (size_t)b * 216 * HW;
    const float fx = flow[(size_t)(b * 2 + 0) * HW + r];
    const float fy = flow[(size_t)(b * 2 + 1) * HW + r];
#pragma unroll
    for (int gi = 0; gi < 2; ++gi) {
      const int g = gq + gi * 4;
#pragma unroll
      for (int k = 0; k < 9; ++k) {
        float dyv = omb[(size_t)(g * 18 + 2 * k) * HW + r] + fy;
        float dxv = omb[(size_t)(g * 18 + 2 * k + 1) * HW + r] + fx;
        float mk = omb[(size_t)(144 + g * 9 + k) * HW + r];
        mk = 1.f / (1.f + __expf(-mk));
        float spy = (float)(y - 1 + k / 3) + dyv;
        float spx = (float)(x - 1 + k % 3) + dxv;
        float y0f = floorf(spy), x0f = floorf(spx);
        float wy = spy - y0f, wx = spx - x0f;
        int y0 = (int)y0f, x0 = (int)x0f, y1 = y0 + 1, x1 = x0 + 1;
        float vy0 = (y0 >= 0 && y0 < HNUM) ? 1.f : 0.f;
        float vy1 = (y1 >= 0 && y1 < HNUM) ? 1.f : 0.f;
        float vx0 = (x0 >= 0 && x0 < WNUM) ? 1.f : 0.f;
        float vx1 = (x1 >= 0 && x1 < WNUM) ? 1.f : 0.f;
        int cy0 = min(max(y0, 0), HNUM - 1), cy1 = min(max(y1, 0), HNUM - 1);
        int cx0 = min(max(x0, 0), WNUM - 1), cx1 = min(max(x1, 0), WNUM - 1);
        float w00 = (1.f - wy) * (1.f - wx) * vy0 * vx0 * mk;
        float w01 = (1.f - wy) * wx * vy0 * vx1 * mk;
        float w10 = wy * (1.f - wx) * vy1 * vx0 * mk;
        float w11 = wy * wx * vy1 * vx1 * mk;
        int i00 = cy0 * WNUM + cx0, i01 = cy0 * WNUM + cx1;
        int i10 = cy1 * WNUM + cx0, i11 = cy1 * WNUM + cx1;
#pragma unroll
        for (int c = 0; c < 8; ++c) {
          const float* im = nbr + (size_t)(b * 64 + g * 8 + c) * HW;
          float v = w00 * im[i00] + w01 * im[i01] + w10 * im[i10] + w11 * im[i11];
          At[p * 576 + (g * 8 + c) * 9 + k] = f2bf(v);
        }
      }
    }
  }
  __syncthreads();
  const int lane = threadIdx.x & 31;
  const int wv = threadIdx.x >> 5;
  const int ocb = wv * 16;
  v8f acc0 = {}, acc1 = {};
  const unsigned short* Arow0 = At + (lane & 15) * 576;
  const unsigned short* Arow1 = At + ((lane & 15) + 16) * 576;
#pragma unroll
  for (int ck = 0; ck < 18; ++ck) {
    v16bf a0 = ldsAFrag(Arow0 + ck * 32, lane);
    v16bf a1 = ldsAFrag(Arow1 + ck * 32, lane);
    v16bf bv = glbBFrag(dcnp, ck, 64, ocb, lane);
    acc0 = __builtin_amdgcn_wmma_f32_16x16x32_bf16(false, a0, false, bv,
                                                   (short)0, acc0, false, false);
    acc1 = __builtin_amdgcn_wmma_f32_16x16x32_bf16(false, a1, false, bv,
                                                   (short)0, acc1, false, false);
  }
  const int oc = ocb + (lane & 15);
  const float bb = bias[oc];
  const int m0 = (lane < 16) ? 0 : 8;
  float* dst = out + ((size_t)(b * 64 + oc) * HNUM + y) * WNUM + xb + m0;
#pragma unroll
  for (int r = 0; r < 8; ++r) {
    dst[r]      = acc0[r] + bb;
    dst[r + 16] = acc1[r] + bb;
  }
}

// ---------------------------------------------------------------- launcher
extern "C" void kernel_launch(void* const* d_in, const int* in_sizes, int n_in,
                              void* d_out, int out_size, void* d_ws, size_t ws_size,
                              hipStream_t stream) {
  const float* nbr  = (const float*)d_in[0];
  const float* ref  = (const float*)d_in[1];
  const float* flow = (const float*)d_in[2];
  const float* w1   = (const float*)d_in[3];
  const float* b1   = (const float*)d_in[4];
  const float* omw  = (const float*)d_in[5];
  const float* omb  = (const float*)d_in[6];
  const float* dcw  = (const float*)d_in[7];
  const float* dcb  = (const float*)d_in[8];
  float* out = (float*)d_out;

  char* ws = (char*)d_ws;
  float* nw  = (float*)ws;  ws += (size_t)BNUM * 64 * HW * 4;    // 33.5 MB
  float* fea = (float*)ws;  ws += (size_t)BNUM * 64 * HW * 4;    // 33.5 MB
  float* om  = (float*)ws;  ws += (size_t)BNUM * 216 * HW * 4;   // 113 MB
  unsigned short* w1p  = (unsigned short*)ws; ws += (size_t)36 * 64 * 32 * 2;
  unsigned short* omp_ = (unsigned short*)ws; ws += (size_t)27 * 256 * 32 * 2;
  unsigned short* dcp  = (unsigned short*)ws; ws += (size_t)18 * 64 * 32 * 2;

  warp_kernel<<<(BNUM * HW + 255) / 256, 256, 0, stream>>>(nbr, flow, nw);
  pack_w1<<<(36 * 64 * 32) / 256, 256, 0, stream>>>(w1, w1p, 36 * 64 * 32);
  pack_om<<<(27 * 256 * 32) / 256, 256, 0, stream>>>(omw, omp_, 27 * 256 * 32);
  pack_dcn<<<(18 * 64 * 32) / 256, 256, 0, stream>>>(dcw, dcp, 18 * 64 * 32);
  conv1_kernel<<<dim3(WNUM / 32, HNUM, BNUM), 128, 0, stream>>>(nw, ref, w1p, b1, fea);
  omconv_kernel<<<dim3((WNUM / 32) * 4, HNUM, BNUM), 128, 0, stream>>>(fea, flow, omp_, omb, om);
  dcn_kernel<<<dim3(WNUM / 32, HNUM, BNUM), 128, 0, stream>>>(nbr, flow, om, dcp, dcb, out);
}
